// LongformerDecoder_69020124446894
// MI455X (gfx1250) — matrix operand order repaired and checked by reading, not compiled, not run on hardware
//
#include <hip/hip_runtime.h>
#include <hip/hip_bf16.h>
#include <math.h>

// ---------------------------------------------------------------------------
// Longformer decoder forward for gfx1250 (MI455X), bf16 WMMA pipeline.
//   B=2 S=2048 V=32000 D=512 H=8 DH=64 NL=4 MLP=2048 WIN/2=256
// ---------------------------------------------------------------------------

#define LB   2
#define LS   2048
#define LV   32000
#define LD   512
#define LH   8
#define LDH  64
#define LNL  4
#define LMLP 2048
#define LM   (LB * LS)        // 4096 rows of activations
#define LWINH 256             // half window
#define LQKV (3 * LD)         // fused qkv row stride (1536)

typedef __attribute__((ext_vector_type(16))) __bf16 v16bf;
typedef __attribute__((ext_vector_type(8)))  float  v8f;
typedef __attribute__((ext_vector_type(4))) unsigned u32x4;
typedef __attribute__((ext_vector_type(8))) int      i32x8;
typedef __attribute__((ext_vector_type(4))) int      i32x4;

union frag16 { uint4 u[2]; v16bf v; __bf16 h[16]; };
union pack2  { __bf16 h[2]; unsigned u; };

static __device__ __forceinline__ v8f wmma_bf16(const v16bf& a, const v16bf& b,
                                                const v8f& c) {
  return __builtin_amdgcn_wmma_f32_16x16x32_bf16(false, a, false, b, (short)0,
                                                 c, false, false);
}

// ---------------------------------------------------------------------------
// Tensor Data Mover (gfx1250): async tile DMA global -> LDS, TENSORcnt-tracked.
// Guarded: falls back to direct-global A-fragment loads if builtin is absent.
// This toolchain exposes the 6-arg form:
//   (u32x4 g0, i32x8 g1, i32x4 g2, i32x4 g3, i32x8 g4, i32 cpol)
// ---------------------------------------------------------------------------
#if __has_builtin(__builtin_amdgcn_tensor_load_to_lds) && \
    __has_builtin(__builtin_amdgcn_s_wait_tensorcnt)
#define LT_HAS_TDM 1
#endif

#ifdef LT_HAS_TDM
// Generic LDS pointer = {aperture_hi32, lds_offset32}: low 32 bits = D#.lds_addr
static __device__ __forceinline__ unsigned lds_addr_of(const void* p) {
  return (unsigned)(unsigned long long)p;
}

// 2D tile load: 32 (x, elements) x 128 (y, rows), 2-byte elements, row-major
// into contiguous LDS (row stride 64 B). D# layout per CDNA5 ISA ch.8.
static __device__ __forceinline__ void tdm_load_a_tile(
    const __bf16* gsrc, unsigned lds_byte_addr, int K) {
  const unsigned long long ga = (unsigned long long)gsrc;
  u32x4 g0;
  g0[0] = 1u;                                  // count=1 (valid descriptor)
  g0[1] = lds_byte_addr;                       // lds_addr  (bits 63:32)
  g0[2] = (unsigned)(ga & 0xFFFFFFFFull);      // global_addr[31:0]
  g0[3] = (unsigned)((ga >> 32) & 0x1FFFFFFull) | (2u << 30);  // hi | type=2
  const unsigned t0 = (unsigned)K;             // tensor_dim0 (elements)
  const unsigned t1 = 0x40000000u;             // tensor_dim1 (huge, no OOB)
  const unsigned s0 = (unsigned)K;             // tensor_dim0_stride (elements)
  i32x8 g1;
  g1[0] = (int)(1u << 16);                     // data_size = 1 -> 2 bytes
  g1[1] = (int)((t0 & 0xFFFFu) << 16);         // tensor_dim0[15:0]
  g1[2] = (int)((t0 >> 16) | ((t1 & 0xFFFFu) << 16));
  g1[3] = (int)((t1 >> 16) | (32u << 16));     // tensor_dim1 hi | tile_dim0=32
  g1[4] = (int)128u;                           // tile_dim1=128, tile_dim2=0
  g1[5] = (int)s0;                             // tensor_dim0_stride[31:0]
  g1[6] = 0;                                   // stride hi / dim1_stride lo
  g1[7] = 0;
  const i32x4 z4 = {0, 0, 0, 0};
  const i32x8 z8 = {0, 0, 0, 0, 0, 0, 0, 0};
  __builtin_amdgcn_tensor_load_to_lds(g0, g1, z4, z4, z8, 0);
}
#endif

// ---------------------------------------------------------------------------
// fp32 -> bf16 weight conversion
// ---------------------------------------------------------------------------
__global__ __launch_bounds__(256) void f32_to_bf16_kernel(
    const float* __restrict__ src, __bf16* __restrict__ dst, size_t n) {
  size_t i = (size_t)blockIdx.x * blockDim.x + threadIdx.x;
  if (i < n) dst[i] = (__bf16)src[i];
}

// pack wq|wk|wv -> per-layer fused [D, 3D] bf16 weight
__global__ __launch_bounds__(256) void pack_qkv_kernel(
    const float* __restrict__ wq, const float* __restrict__ wk,
    const float* __restrict__ wv, __bf16* __restrict__ dst) {
  const size_t per = (size_t)LNL * LD * LD;
  size_t i = (size_t)blockIdx.x * blockDim.x + threadIdx.x;
  if (i >= 3 * per) return;
  const int sel = (int)(i / per);
  const size_t r = i % per;
  const int l   = (int)(r / (LD * LD));
  const int rem = (int)(r % (LD * LD));
  const int row = rem / LD;
  const int col = rem % LD;
  const float* src = (sel == 0) ? wq : (sel == 1) ? wk : wv;
  dst[((size_t)l * LD + row) * LQKV + sel * LD + col] = (__bf16)src[r];
}

// ---------------------------------------------------------------------------
// embedding gather (shift-right) + sinusoidal positional encoding -> x (fp32)
// ---------------------------------------------------------------------------
__global__ __launch_bounds__(128) void embed_pe_kernel(
    const int* __restrict__ ids, const float* __restrict__ embed,
    float* __restrict__ x) {
  const int bs = blockIdx.x;      // b*S + s
  const int s  = bs % LS;
  const int b  = bs / LS;
  const int id = (s == 0) ? 0 : ids[b * LS + s - 1];
  const float* e = embed + (size_t)id * LD;
  const float c = -logf(10000.0f) / (float)LD;
  for (int i = threadIdx.x; i < LD; i += 128) {
    const int pair = i >> 1;
    const float ang = (float)s * __expf(c * (float)(2 * pair));
    const float pe = (i & 1) ? __cosf(ang) : __sinf(ang);
    x[(size_t)bs * LD + i] = e[i] + pe;
  }
}

// ---------------------------------------------------------------------------
// LayerNorm: fp32 x -> bf16 y   (one row of width D per 128-thread block)
// ---------------------------------------------------------------------------
__global__ __launch_bounds__(128) void layernorm_bf16_kernel(
    const float* __restrict__ x, const float* __restrict__ sc,
    const float* __restrict__ bi, __bf16* __restrict__ y) {
  __shared__ float red[2][4];
  const int row = blockIdx.x;
  const float* xr = x + (size_t)row * LD;
  float sum = 0.f, sq = 0.f;
  for (int i = threadIdx.x; i < LD; i += 128) {
    const float v = xr[i];
    sum += v; sq += v * v;
  }
  for (int off = 16; off; off >>= 1) {
    sum += __shfl_xor(sum, off, 32);
    sq  += __shfl_xor(sq,  off, 32);
  }
  const int lane = threadIdx.x & 31, wv = threadIdx.x >> 5;
  if (lane == 0) { red[0][wv] = sum; red[1][wv] = sq; }
  __syncthreads();
  sum = red[0][0] + red[0][1] + red[0][2] + red[0][3];
  sq  = red[1][0] + red[1][1] + red[1][2] + red[1][3];
  const float mu = sum / (float)LD;
  const float var = sq / (float)LD - mu * mu;
  const float rstd = rsqrtf(var + 1e-6f);
  for (int i = threadIdx.x; i < LD; i += 128) {
    const float v = (xr[i] - mu) * rstd * sc[i] + bi[i];
    y[(size_t)row * LD + i] = (__bf16)v;
  }
}

// ---------------------------------------------------------------------------
// bf16 WMMA GEMM: C[M,N] = A[M,K] * W[K,N] (+bias)(+gelu)(+residual)
// 128(M)x128(N) tile per 128-thread block; 4 waves, each owns 32(M)x128(N).
// A tile: TDM (tensor_load_to_lds) double-buffered into row-major LDS;
//         fragments are 2x ds_load_b128 per lane. (Fallback: direct global.)
// B tile: register-pipelined global loads, committed transposed to LDS
//         (Bt[col][k]); fragments are 2x ds_load_b128.
// ---------------------------------------------------------------------------
template <bool OUT_BF16, bool BIAS, bool RES, bool GELU>
__global__ __launch_bounds__(128) void gemm_bf16_kernel(
    const __bf16* __restrict__ A, const __bf16* __restrict__ W,
    const float* __restrict__ bias, const float* __restrict__ res,
    void* __restrict__ Cout, int N, int K) {
  __shared__ __attribute__((aligned(16))) __bf16 Bt[128][40];  // 80-B col rows
#ifdef LT_HAS_TDM
  __shared__ __attribute__((aligned(16))) __bf16 At[2][128][32];  // row-major
#endif
  const int tid  = threadIdx.x;
  const int lane = tid & 31;
  const int wave = tid >> 5;
  const int m0 = blockIdx.x * 128;
  const int n0 = blockIdx.y * 128;
  const int l16  = lane & 15;
  const int half = lane >> 4;

  // staging assignment: k-row pair + 16-col octet per thread
  const int rp  = tid >> 3;   // 0..15  -> k rows 2rp, 2rp+1
  const int oct = tid & 7;    // cols 16*oct .. 16*oct+15

  const __bf16* wptr = W + (size_t)(2 * rp) * N + n0 + 16 * oct;
#ifdef LT_HAS_TDM
  const __bf16* agbase = A + (size_t)m0 * K;   // tile row base (rows m0..m0+127)
#else
  const __bf16* aptr0 = A + (size_t)(m0 + 32 * wave + l16) * K + half * 8;
  const __bf16* aptr1 = aptr0 + (size_t)16 * K;
#endif

  v8f acc[2][8];
#pragma unroll
  for (int mi = 0; mi < 2; ++mi)
#pragma unroll
    for (int i = 0; i < 8; ++i) acc[mi][i] = v8f{};

  // prologue: first B tile into registers (+ first A tile via TDM)
  frag16 r0, r1;
  r0.u[0] = *(const uint4*)wptr;       r0.u[1] = *(const uint4*)(wptr + 8);
  r1.u[0] = *(const uint4*)(wptr + N); r1.u[1] = *(const uint4*)(wptr + N + 8);
#ifdef LT_HAS_TDM
  if (wave == 0) tdm_load_a_tile(agbase, lds_addr_of(&At[0][0][0]), K);
  int abuf = 0;
#endif

  for (int kc = 0; kc < K; kc += 32) {
    // ---- commit staged B tile to LDS, transposed ----
#pragma unroll
    for (int i = 0; i < 16; ++i) {
      pack2 p;
      p.h[0] = r0.h[i];
      p.h[1] = r1.h[i];
      *(unsigned*)&Bt[16 * oct + i][2 * rp] = p.u;
    }
#ifdef LT_HAS_TDM
    // issue next A tile into alternate buffer; drain current (in-order TDM)
    if (wave == 0) {
      if (kc + 32 < K) {
        tdm_load_a_tile(agbase + (kc + 32),
                        lds_addr_of(&At[abuf ^ 1][0][0]), K);
        __builtin_amdgcn_s_wait_tensorcnt((short)1);
      } else {
        __builtin_amdgcn_s_wait_tensorcnt((short)0);
      }
    }
#endif
    __syncthreads();

    // ---- issue next B tile's global loads (latency hidden by WMMAs) ----
    if (kc + 32 < K) {
      const __bf16* wn = wptr + (size_t)(kc + 32) * N;
      r0.u[0] = *(const uint4*)wn;       r0.u[1] = *(const uint4*)(wn + 8);
      r1.u[0] = *(const uint4*)(wn + N); r1.u[1] = *(const uint4*)(wn + N + 8);
    }

    // ---- A fragments: two contiguous 8-elem runs ----
    frag16 ta0, ta1;
#ifdef LT_HAS_TDM
    const __bf16* arow0 = &At[abuf][32 * wave + l16][half * 8];
    const __bf16* arow1 = arow0 + 16 * 32;
    ta0.u[0] = *(const uint4*)arow0;
    ta0.u[1] = *(const uint4*)(arow0 + 16);
    ta1.u[0] = *(const uint4*)arow1;
    ta1.u[1] = *(const uint4*)(arow1 + 16);
    abuf ^= 1;
#else
    ta0.u[0] = *(const uint4*)(aptr0 + kc);
    ta0.u[1] = *(const uint4*)(aptr0 + kc + 16);
    ta1.u[0] = *(const uint4*)(aptr1 + kc);
    ta1.u[1] = *(const uint4*)(aptr1 + kc + 16);
#endif

#pragma unroll
    for (int nt = 0; nt < 8; ++nt) {
      frag16 tb;
      const __bf16* bp = &Bt[16 * nt + l16][8 * half];
      tb.u[0] = *(const uint4*)bp;
      tb.u[1] = *(const uint4*)(bp + 16);
      acc[0][nt] = wmma_bf16(ta0.v, tb.v, acc[0][nt]);
      acc[1][nt] = wmma_bf16(ta1.v, tb.v, acc[1][nt]);
    }
    __syncthreads();
  }

  // epilogue: C element r -> row = r + 8*half, col = 16*nt + lane%16
#pragma unroll
  for (int mi = 0; mi < 2; ++mi) {
    const int crow0 = m0 + 32 * wave + 16 * mi + 8 * half;
#pragma unroll
    for (int nt = 0; nt < 8; ++nt) {
      const int col = n0 + 16 * nt + l16;
      const float bval = BIAS ? bias[col] : 0.0f;
#pragma unroll
      for (int r = 0; r < 8; ++r) {
        const size_t idx = (size_t)(crow0 + r) * N + col;
        float v = acc[mi][nt][r] + bval;
        if (GELU) {
          const float u = 0.7978845608028654f * (v + 0.044715f * v * v * v);
          v = 0.5f * v * (1.0f + tanhf(u));
        }
        if (RES) v += res[idx];
        if (OUT_BF16) ((__bf16*)Cout)[idx] = (__bf16)v;
        else          ((float*)Cout)[idx]  = v;
      }
    }
  }
}

// ---------------------------------------------------------------------------
// Sliding-window causal attention, flash-style streaming softmax.
// One wave per (b, h, 16-row query tile); keys walked in 32-aligned blocks.
// Q/K/V come from the fused qkv buffer (row stride 1536, col ofs 0/512/1024).
// Scores: Q(16x64) x K^T -> two 16x16 WMMA tiles per block.
// P(16x32, full K) @ V(32x64): P and V staged transposed in wave-private LDS,
// every fragment is 2x ds_load_b128; V global reads are 8x b128 per block.
// ---------------------------------------------------------------------------
__global__ __launch_bounds__(32) void attn_window_kernel(
    const __bf16* __restrict__ qkv, const int* __restrict__ inputs,
    __bf16* __restrict__ att, float scale) {
  __shared__ __attribute__((aligned(16))) __bf16 Pt[16][40];  // probs 16x32
  __shared__ __attribute__((aligned(16))) __bf16 Vt[64][40];  // V^T 64x32
  const int lane = threadIdx.x & 31;
  const int qt = blockIdx.x;   // S/16
  const int h  = blockIdx.y;
  const int b  = blockIdx.z;
  const int l16  = lane & 15;
  const int half = lane >> 4;

  // Q fragments (A layout), 2 chunks of K=32 covering DH=64
  v16bf qa[2];
  {
    const __bf16* qp =
        qkv + (size_t)(b * LS + qt * 16 + l16) * LQKV + h * LDH + half * 8;
#pragma unroll
    for (int c2 = 0; c2 < 2; ++c2) {
      frag16 t;
      t.u[0] = *(const uint4*)(qp + c2 * 32);
      t.u[1] = *(const uint4*)(qp + c2 * 32 + 16);
      qa[c2] = t.v;
    }
  }
  int padq[8];
#pragma unroll
  for (int r = 0; r < 8; ++r)
    padq[r] = inputs[b * LS + qt * 16 + r + 8 * half] > 0;

  float m[8], lsum[8];
  v8f acc[4] = {v8f{}, v8f{}, v8f{}, v8f{}};
#pragma unroll
  for (int r = 0; r < 8; ++r) { m[r] = -1e30f; lsum[r] = 0.0f; }

  const int t0 = qt * 16 - LWINH;
  const int jb0 = (t0 <= 0) ? 0 : (t0 >> 5);
  const int jb1 = (qt * 16 + 15) >> 5;

  for (int jb = jb0; jb <= jb1; ++jb) {
    const int kbase = jb * 32;
    // ---- scores: two 16-key subtiles, 2 WMMAs each over DH=64 ----
    v8f c[2] = {v8f{}, v8f{}};
#pragma unroll
    for (int t = 0; t < 2; ++t) {
      const __bf16* kp = qkv +
          (size_t)(b * LS + kbase + t * 16 + l16) * LQKV + LD + h * LDH +
          half * 8;
#pragma unroll
      for (int c2 = 0; c2 < 2; ++c2) {
        frag16 tb;
        tb.u[0] = *(const uint4*)(kp + c2 * 32);
        tb.u[1] = *(const uint4*)(kp + c2 * 32 + 16);
        c[t] = wmma_bf16(qa[c2], tb.v, c[t]);
      }
    }
    const int padk0 = inputs[b * LS + kbase + l16] > 0;
    const int padk1 = inputs[b * LS + kbase + 16 + l16] > 0;

    // ---- streaming softmax over 32 columns ----
    float p0[8], p1[8];
#pragma unroll
    for (int r = 0; r < 8; ++r) {
      const int qrow = qt * 16 + r + 8 * half;
      const int key0 = kbase + l16;
      const int key1 = key0 + 16;
      const bool ok0 =
          (key0 <= qrow) && (qrow - key0 <= LWINH) && padk0 && padq[r];
      const bool ok1 =
          (key1 <= qrow) && (qrow - key1 <= LWINH) && padk1 && padq[r];
      const float s0 = ok0 ? c[0][r] * scale : -1e9f;
      const float s1 = ok1 ? c[1][r] * scale : -1e9f;
      float tm = fmaxf(s0, s1);
      tm = fmaxf(tm, __shfl_xor(tm, 1, 32));
      tm = fmaxf(tm, __shfl_xor(tm, 2, 32));
      tm = fmaxf(tm, __shfl_xor(tm, 4, 32));
      tm = fmaxf(tm, __shfl_xor(tm, 8, 32));
      const float mn = fmaxf(m[r], tm);
      const float alpha = __expf(m[r] - mn);
      m[r] = mn;
      p0[r] = __expf(s0 - mn);
      p1[r] = __expf(s1 - mn);
      float ps = p0[r] + p1[r];
      ps += __shfl_xor(ps, 1, 32);
      ps += __shfl_xor(ps, 2, 32);
      ps += __shfl_xor(ps, 4, 32);
      ps += __shfl_xor(ps, 8, 32);
      lsum[r] = lsum[r] * alpha + ps;
#pragma unroll
      for (int nt = 0; nt < 4; ++nt) acc[nt][r] *= alpha;
    }
    __syncthreads();
    // probs -> LDS (row-major 16x32)
#pragma unroll
    for (int r = 0; r < 8; ++r) {
      Pt[r + 8 * half][l16]      = (__bf16)p0[r];
      Pt[r + 8 * half][16 + l16] = (__bf16)p1[r];
    }
    // V block 32x64 -> LDS transposed: lane owns key pair (2*l16) x 32-dh half
    {
      const __bf16* vp0 = qkv +
          (size_t)(b * LS + kbase + 2 * l16) * LQKV + 2 * LD + h * LDH +
          32 * half;
      const __bf16* vp1 = vp0 + LQKV;
      union { uint4 u[4]; __bf16 h[32]; } w0, w1;
#pragma unroll
      for (int i = 0; i < 4; ++i) {
        w0.u[i] = *(const uint4*)(vp0 + 8 * i);
        w1.u[i] = *(const uint4*)(vp1 + 8 * i);
      }
#pragma unroll
      for (int i = 0; i < 32; ++i) {
        pack2 p;
        p.h[0] = w0.h[i];
        p.h[1] = w1.h[i];
        *(unsigned*)&Vt[32 * half + i][2 * l16] = p.u;
      }
    }
    __syncthreads();
    // P fragment (A layout, K=32 full)
    frag16 tp;
    {
      const __bf16* pp = &Pt[l16][half * 8];
      tp.u[0] = *(const uint4*)pp;
      tp.u[1] = *(const uint4*)(pp + 16);
    }
#pragma unroll
    for (int nt = 0; nt < 4; ++nt) {
      frag16 tv;
      const __bf16* vp = &Vt[16 * nt + l16][8 * half];
      tv.u[0] = *(const uint4*)vp;
      tv.u[1] = *(const uint4*)(vp + 16);
      acc[nt] = wmma_bf16(tp.v, tv.v, acc[nt]);
    }
    __syncthreads();
  }
  // normalize + store bf16 att[b,s,h*DH+dh]
#pragma unroll
  for (int r = 0; r < 8; ++r) {
    const int qrow = qt * 16 + r + 8 * half;
    const float inv = 1.0f / lsum[r];
#pragma unroll
    for (int nt = 0; nt < 4; ++nt)
      att[(size_t)(b * LS + qrow) * LD + h * LDH + nt * 16 + l16] =
          (__bf16)(acc[nt][r] * inv);
  }
}

// ---------------------------------------------------------------------------
// host launcher
// ---------------------------------------------------------------------------
static inline size_t align256(size_t v) { return (v + 255) & ~(size_t)255; }

extern "C" void kernel_launch(void* const* d_in, const int* in_sizes, int n_in,
                              void* d_out, int out_size, void* d_ws, size_t ws_size,
                              hipStream_t stream) {
  (void)in_sizes; (void)n_in; (void)out_size; (void)ws_size;
  const int*   inputs = (const int*)d_in[0];
  const float* embed  = (const float*)d_in[1];
  const float* wq     = (const float*)d_in[2];
  const float* wk     = (const float*)d_in[3];
  const float* wv     = (const float*)d_in[4];
  const float* wo     = (const float*)d_in[5];
  const float* ln1_s  = (const float*)d_in[6];
  const float* ln1_b  = (const float*)d_in[7];
  const float* ln2_s  = (const float*)d_in[8];
  const float* ln2_b  = (const float*)d_in[9];
  const float* w1     = (const float*)d_in[10];
  const float* b1     = (const float*)d_in[11];
  const float* w2     = (const float*)d_in[12];
  const float* b2     = (const float*)d_in[13];
  const float* lnf_s  = (const float*)d_in[14];
  const float* lnf_b  = (const float*)d_in[15];
  const float* w_out  = (const float*)d_in[16];
  const float* b_out  = (const float*)d_in[17];
  float* out = (float*)d_out;

  // workspace carve-up
  char* ws = (char*)d_ws;
  size_t off = 0;
  auto take = [&](size_t bytes) { char* p = ws + off; off += align256(bytes); return p; };
  float*  x     = (float*)take((size_t)LM * LD * 4);
  __bf16* y     = (__bf16*)take((size_t)LM * LD * 2);
  __bf16* qkvb  = (__bf16*)take((size_t)LM * LQKV * 2);
  __bf16* attb  = (__bf16*)take((size_t)LM * LD * 2);
  __bf16* hb    = (__bf16*)take((size_t)LM * LMLP * 2);
  __bf16* qkvwb = (__bf16*)take((size_t)LNL * LD * LQKV * 2);
  __bf16* wob   = (__bf16*)take((size_t)LNL * LD * LD * 2);
  __bf16* w1b   = (__bf16*)take((size_t)LNL * LD * LMLP * 2);
  __bf16* w2b   = (__bf16*)take((size_t)LNL * LMLP * LD * 2);
  __bf16* wOb   = (__bf16*)take((size_t)LD * LV * 2);

  auto conv = [&](const float* s, __bf16* d, size_t n) {
    f32_to_bf16_kernel<<<dim3((unsigned)((n + 255) / 256)), dim3(256), 0, stream>>>(s, d, n);
  };
  pack_qkv_kernel<<<dim3((unsigned)((3 * (size_t)LNL * LD * LD + 255) / 256)),
                   dim3(256), 0, stream>>>(wq, wk, wv, qkvwb);
  conv(wo, wob, (size_t)LNL * LD * LD);
  conv(w1, w1b, (size_t)LNL * LD * LMLP);
  conv(w2, w2b, (size_t)LNL * LMLP * LD);
  conv(w_out, wOb, (size_t)LD * LV);

  embed_pe_kernel<<<dim3(LM), dim3(128), 0, stream>>>(inputs, embed, x);

  const dim3 gemmBlk(128);
  const float scale = 1.0f / sqrtf((float)LDH);

  for (int l = 0; l < LNL; ++l) {
    layernorm_bf16_kernel<<<dim3(LM), dim3(128), 0, stream>>>(
        x, ln1_s + l * LD, ln1_b + l * LD, y);
    // fused Q|K|V projection (bf16 out, no bias), N = 1536
    gemm_bf16_kernel<true, false, false, false>
        <<<dim3(LM / 128, LQKV / 128), gemmBlk, 0, stream>>>(
            y, qkvwb + (size_t)l * LD * LQKV, nullptr, nullptr, qkvb, LQKV, LD);
    // sliding-window attention
    attn_window_kernel<<<dim3(LS / 16, LH, LB), dim3(32), 0, stream>>>(
        qkvb, inputs, attb, scale);
    // output projection + residual into x (fp32)
    gemm_bf16_kernel<false, false, true, false>
        <<<dim3(LM / 128, LD / 128), gemmBlk, 0, stream>>>(
            attb, wob + (size_t)l * LD * LD, nullptr, x, x, LD, LD);
    // MLP
    layernorm_bf16_kernel<<<dim3(LM), dim3(128), 0, stream>>>(
        x, ln2_s + l * LD, ln2_b + l * LD, y);
    gemm_bf16_kernel<true, true, false, true>
        <<<dim3(LM / 128, LMLP / 128), gemmBlk, 0, stream>>>(
            y, w1b + (size_t)l * LD * LMLP, b1 + l * LMLP, nullptr, hb, LMLP, LD);
    gemm_bf16_kernel<false, true, true, false>
        <<<dim3(LM / 128, LD / 128), gemmBlk, 0, stream>>>(
            hb, w2b + (size_t)l * LMLP * LD, b2 + l * LD, x, x, LD, LMLP);
  }

  // final LN + vocab projection (fp32 logits)
  layernorm_bf16_kernel<<<dim3(LM), dim3(128), 0, stream>>>(x, lnf_s, lnf_b, y);
  gemm_bf16_kernel<false, true, false, false>
      <<<dim3(LM / 128, LV / 128), gemmBlk, 0, stream>>>(
          y, wOb, b_out, nullptr, out, LV, LD);
}
